// DynamicSkippingMixtralSparseMoeBlockWrapper_5420248728297
// MI455X (gfx1250) — matrix operand order repaired and checked by reading, not compiled, not run on hardware
//
#include <hip/hip_runtime.h>
#include <hip/hip_bf16.h>
#include <stdint.h>

// ---------------------------------------------------------------------------
// MoE (Mixtral-style, top-2 with dynamic skipping) for MI455X / gfx1250.
// Routed gather-FFN-scatter; bf16 WMMA (v_wmma_f32_16x16x32_bf16) with
// on-the-fly fp32->bf16 weight conversion in the staging path (pre-converting
// weights would double HBM traffic: 352MB -> 704MB). fp32 accumulate, fp32 out.
// Pure-copy LDS fills (bf16 activations / h tiles) use the gfx1250 async
// global->LDS path (ASYNCcnt) when the builtin is available.
// ---------------------------------------------------------------------------

#define T_TOKENS 4096
#define H_DIM    1024
#define F_DIM    3584
#define NEXP     8
#define BETA_F   0.5f

typedef __attribute__((ext_vector_type(16))) __bf16       v16bf;
typedef __attribute__((ext_vector_type(8)))  float        v8f;
typedef __attribute__((ext_vector_type(4)))  unsigned int u32x4;
typedef __attribute__((ext_vector_type(4)))  float        f32x4;
typedef __attribute__((ext_vector_type(4)))  int          i32x4;

union Frag { u32x4 u[2]; v16bf v; };

#if defined(__AMDGCN__) && \
    __has_builtin(__builtin_amdgcn_global_load_async_to_lds_b128) && \
    __has_builtin(__builtin_amdgcn_global_load_async_to_lds_b32)
#define ASYNC_CP 1
// Address-space-qualified pointer typedefs. AS1/AS3-qualified pointers
// implicitly convert to generic, so these work whether the builtin takes
// AS-qualified or generic pointers.
typedef __attribute__((address_space(1))) i32x4 gas_i32x4;
typedef __attribute__((address_space(3))) i32x4 las_i32x4;
typedef __attribute__((address_space(1))) int   gas_i32;
typedef __attribute__((address_space(3))) int   las_i32;
#else
#define ASYNC_CP 0
#endif

// 16-byte global -> LDS copy (async when supported)
__device__ __forceinline__ void cp16_to_lds(const void* g, void* l) {
#if ASYNC_CP
  void* gv = const_cast<void*>(g);
  __builtin_amdgcn_global_load_async_to_lds_b128(
      (gas_i32x4*)gv, (las_i32x4*)l, 0, 0);
#else
  *(u32x4*)l = *(const u32x4*)g;
#endif
}
// 4-byte global -> LDS copy (async when supported)
__device__ __forceinline__ void cp4_to_lds(const void* g, void* l) {
#if ASYNC_CP
  void* gv = const_cast<void*>(g);
  __builtin_amdgcn_global_load_async_to_lds_b32(
      (gas_i32*)gv, (las_i32*)l, 0, 0);
#else
  *(unsigned int*)l = *(const unsigned int*)g;
#endif
}
__device__ __forceinline__ void cp_wait() {
#if ASYNC_CP
#if __has_builtin(__builtin_amdgcn_s_wait_asynccnt)
  __builtin_amdgcn_s_wait_asynccnt(0);
#else
  asm volatile("s_wait_asynccnt 0" ::: "memory");
#endif
#endif
}

__device__ __forceinline__ unsigned int f2bf1(float f) {
  unsigned int u = __builtin_bit_cast(unsigned int, f);
  return (u + 0x7FFFu + ((u >> 16) & 1u)) >> 16;   // RNE truncation to bf16
}
__device__ __forceinline__ unsigned int f2bf2(float lo, float hi) {
  return f2bf1(lo) | (f2bf1(hi) << 16);
}

// ---------------------------------------------------------------------------
// Router: one wave per token. Logits (written to output tail), fp32 softmax,
// top-2 + dynamic skip + renorm, expert histogram, bf16 activation copy.
// ---------------------------------------------------------------------------
__global__ __launch_bounds__(256)
void router_kernel(const float* __restrict__ x, const float* __restrict__ gw,
                   float* __restrict__ logits_out, unsigned short* __restrict__ xb,
                   int* __restrict__ counts, int* __restrict__ sel,
                   float* __restrict__ wts)
{
  const int wave = threadIdx.x >> 5;
  const int lane = threadIdx.x & 31;
  const int t = blockIdx.x * 8 + wave;

  const float* xr = x + (size_t)t * H_DIM;
  float xv[32];
#pragma unroll
  for (int i = 0; i < 32; ++i) xv[i] = xr[lane + 32 * i];

  float le[NEXP];
#pragma unroll
  for (int e = 0; e < NEXP; ++e) {
    const float* gr = gw + (size_t)e * H_DIM;
    float acc = 0.f;
#pragma unroll
    for (int i = 0; i < 32; ++i) acc += xv[i] * gr[lane + 32 * i];
#pragma unroll
    for (int off = 16; off > 0; off >>= 1) acc += __shfl_down(acc, off, 32);
    le[e] = acc;                         // full sum valid in lane 0
  }

#pragma unroll
  for (int i = 0; i < 32; ++i)
    xb[(size_t)t * H_DIM + lane + 32 * i] = (unsigned short)f2bf1(xv[i]);

  if (lane == 0) {
#pragma unroll
    for (int e = 0; e < NEXP; ++e) logits_out[t * NEXP + e] = le[e];

    float m = le[0];
#pragma unroll
    for (int e = 1; e < NEXP; ++e) m = fmaxf(m, le[e]);
    float p[NEXP], s = 0.f;
#pragma unroll
    for (int e = 0; e < NEXP; ++e) { p[e] = __expf(le[e] - m); s += p[e]; }

    int i0 = 0;
#pragma unroll
    for (int e = 1; e < NEXP; ++e) if (p[e] > p[i0]) i0 = e;
    int i1 = (i0 == 0) ? 1 : 0;
#pragma unroll
    for (int e = 0; e < NEXP; ++e) if (e != i0 && p[e] > p[i1]) i1 = e;

    float p0 = p[i0] / s, p1 = p[i1] / s;
    float w0, w1; int s1 = i1;
    if (p1 < BETA_F * p0) { w0 = 1.f; w1 = 0.f; s1 = -1; }   // dynamic skip
    else { float d = p0 + p1; w0 = p0 / d; w1 = p1 / d; }

    sel[t * 2 + 0] = i0;  sel[t * 2 + 1] = s1;
    wts[t * 2 + 0] = w0;  wts[t * 2 + 1] = w1;
    atomicAdd(&counts[i0], 1);
    if (s1 >= 0) atomicAdd(&counts[s1], 1);
  }
}

// ---------------------------------------------------------------------------
// Deterministic compaction: prefix offsets + ballot/popc scan in token order.
// ---------------------------------------------------------------------------
__global__ __launch_bounds__(256)
void scan_kernel(const int* __restrict__ counts, int* __restrict__ offsets,
                 const int* __restrict__ sel, const float* __restrict__ wts,
                 int* __restrict__ tok_idx, float* __restrict__ tok_w)
{
  __shared__ int offs[NEXP + 1];
  if (threadIdx.x == 0) {
    int a = 0;
    for (int e = 0; e < NEXP; ++e) { offs[e] = a; a += counts[e]; }
    offs[NEXP] = a;
    for (int e = 0; e <= NEXP; ++e) offsets[e] = offs[e];
  }
  __syncthreads();

  const int e = threadIdx.x >> 5;        // 8 waves == 8 experts
  const int lane = threadIdx.x & 31;
  int base = offs[e];
  for (int t0 = 0; t0 < T_TOKENS; t0 += 32) {
    const int t = t0 + lane;
    const int s0 = sel[t * 2 + 0], s1 = sel[t * 2 + 1];
    const bool hit = (s0 == e) || (s1 == e);
    const float w = (s0 == e) ? wts[t * 2 + 0] : wts[t * 2 + 1];
    const unsigned int m32 = (unsigned int)__ballot(hit);
    if (hit) {
      const int pos = base + __popc(m32 & ((1u << lane) - 1u));
      tok_idx[pos] = t;
      tok_w[pos]   = w;
    }
    base += __popc(m32);
  }
}

// ---------------------------------------------------------------------------
// Up projection: h = silu(x @ w1^T) * (x @ w3^T). Block = 16 tokens x 128 F
// columns, 8 waves each own a 16x16 sub-tile; K over H in 32-steps, one
// v_wmma_f32_16x16x32_bf16 per matrix per step.
// ---------------------------------------------------------------------------
__global__ __launch_bounds__(256)
void moe_up_kernel(const unsigned short* __restrict__ xb,
                   const float* __restrict__ w1, const float* __restrict__ w3,
                   const int* __restrict__ offsets,
                   const int* __restrict__ tok_idx,
                   unsigned short* __restrict__ hbuf)
{
  __shared__ unsigned int xa[16][512];     // 16 tokens x 1024 bf16 (32 KB)
  __shared__ unsigned int w1s[128][16];    // 128 f-rows x 32 k bf16 (8 KB)
  __shared__ unsigned int w3s[128][16];    // 8 KB
  __shared__ int tokL[16];

  const int e = blockIdx.x >> 8;
  const int tile = blockIdx.x & 255;
  const int sBeg = offsets[e], sEnd = offsets[e + 1];
  const int slotBase = sBeg + tile * 16;
  if (slotBase >= sEnd) return;            // uniform early-out
  const int fb = blockIdx.y * 128;

  const int tid = threadIdx.x;
  if (tid < 16) {
    const int s = slotBase + tid;
    tokL[tid] = tok_idx[s < sEnd ? s : slotBase];   // clamp pad rows
  }
  __syncthreads();

  // Gather 16 bf16 activation rows into LDS: 2048 x 16B (async when possible)
#pragma unroll
  for (int i = 0; i < 8; ++i) {
    const int idx = tid + 256 * i;
    const int r = idx >> 7, c = idx & 127;       // 128 x b128 per row
    const unsigned int* src =
        (const unsigned int*)(xb + (size_t)tokL[r] * H_DIM) + c * 4;
    cp16_to_lds(src, &xa[r][c * 4]);
  }
  cp_wait();
  __syncthreads();

  const float* w1e = w1 + (size_t)e * F_DIM * H_DIM;
  const float* w3e = w3 + (size_t)e * F_DIM * H_DIM;
  const int wave = tid >> 5, lane = tid & 31;
  const int g = lane >> 4, mr = lane & 15;

  // Hoisted per-thread staging bases: thread covers rows (tid>>3)+32*it,
  // 4 consecutive floats at column (tid&7)*4; advance by 32 K per step.
  const float* pw1 = w1e + (size_t)(fb + (tid >> 3)) * H_DIM + (tid & 7) * 4;
  const float* pw3 = w3e + (size_t)(fb + (tid >> 3)) * H_DIM + (tid & 7) * 4;
  unsigned int* s1p = &w1s[tid >> 3][(tid & 7) * 2];
  unsigned int* s3p = &w3s[tid >> 3][(tid & 7) * 2];

  v8f c1 = {}; v8f c3 = {};
  for (int k0 = 0; k0 < H_DIM; k0 += 32) {
#pragma unroll
    for (int it = 0; it < 4; ++it) {
      const f32x4 a = *(const f32x4*)(pw1 + (size_t)(32 * it) * H_DIM);
      const f32x4 b = *(const f32x4*)(pw3 + (size_t)(32 * it) * H_DIM);
      s1p[it * 512 + 0] = f2bf2(a[0], a[1]);
      s1p[it * 512 + 1] = f2bf2(a[2], a[3]);
      s3p[it * 512 + 0] = f2bf2(b[0], b[1]);
      s3p[it * 512 + 1] = f2bf2(b[2], b[3]);
    }
    pw1 += 32; pw3 += 32;
    if (k0 + 32 < H_DIM) __builtin_prefetch(pw1, 0, 1);
    __syncthreads();

    Frag A, B1, B3;
    // A (16x32 bf16): lanes 0-15 K{0-15}, lanes 16-31 K{16-31}
    A.u[0] = *(const u32x4*)&xa[mr][(k0 >> 1) + g * 8];
    A.u[1] = *(const u32x4*)&xa[mr][(k0 >> 1) + g * 8 + 4];
    const int br = wave * 16 + mr;               // B column = lane&15
    B1.u[0] = *(const u32x4*)&w1s[br][g * 8];
    B1.u[1] = *(const u32x4*)&w1s[br][g * 8 + 4];
    B3.u[0] = *(const u32x4*)&w3s[br][g * 8];
    B3.u[1] = *(const u32x4*)&w3s[br][g * 8 + 4];

    c1 = __builtin_amdgcn_wmma_f32_16x16x32_bf16(false, A.v, false, B1.v,
                                                 (short)0, c1, false, false);
    c3 = __builtin_amdgcn_wmma_f32_16x16x32_bf16(false, A.v, false, B3.v,
                                                 (short)0, c3, false, false);
    __syncthreads();
  }

  // SiLU(c1) * c3 -> bf16 h buffer. C layout: VGPR i -> M = i + 8*(lane>=16).
#pragma unroll
  for (int i = 0; i < 8; ++i) {
    const float a = c1[i];
    const float hv = (a / (1.f + __expf(-a))) * c3[i];
    const int m = g * 8 + i;
    const size_t row = (size_t)(slotBase + m);
    hbuf[row * F_DIM + fb + wave * 16 + mr] = (unsigned short)f2bf1(hv);
  }
}

// ---------------------------------------------------------------------------
// Down projection: y[token] += routing_w * (h @ w2^T). Block = 16 tokens x
// 128 H-columns; K over F (112 WMMA steps). Scatter via fp32 atomic add onto
// zeroed y (<=2 contributions per element -> order-independent result).
// ---------------------------------------------------------------------------
__global__ __launch_bounds__(256)
void moe_down_kernel(const unsigned short* __restrict__ hbuf,
                     const float* __restrict__ w2,
                     const int* __restrict__ offsets,
                     const int* __restrict__ tok_idx,
                     const float* __restrict__ tok_w,
                     float* __restrict__ y)
{
  __shared__ unsigned int As[16][16];      // 16 rows x 32 k bf16 (1 KB)
  __shared__ unsigned int w2s[128][16];    // 128 n-rows x 32 k bf16 (8 KB)

  const int e = blockIdx.x >> 8;
  const int tile = blockIdx.x & 255;
  const int sBeg = offsets[e], sEnd = offsets[e + 1];
  const int slotBase = sBeg + tile * 16;
  if (slotBase >= sEnd) return;
  const int nb = blockIdx.y * 128;

  const float* w2e = w2 + (size_t)e * H_DIM * F_DIM;
  const int tid = threadIdx.x;
  const int wave = tid >> 5, lane = tid & 31;
  const int g = lane >> 4, mr = lane & 15;

  // Hoisted staging bases.
  const unsigned short* hp =
      hbuf + (size_t)(slotBase + (tid >> 4)) * F_DIM + (tid & 15) * 2;
  unsigned int* asp = &As[tid >> 4][tid & 15];
  const float* pw2 = w2e + (size_t)(nb + (tid >> 3)) * F_DIM + (tid & 7) * 4;
  unsigned int* s2p = &w2s[tid >> 3][(tid & 7) * 2];

  v8f acc = {};
  for (int k0 = 0; k0 < F_DIM; k0 += 32) {
    cp4_to_lds(hp, asp);                   // A tile: 16 rows x 32 k (bf16)
    hp += 32;
#pragma unroll
    for (int it = 0; it < 4; ++it) {       // w2 slab fp32 -> bf16
      const f32x4 v = *(const f32x4*)(pw2 + (size_t)(32 * it) * F_DIM);
      s2p[it * 512 + 0] = f2bf2(v[0], v[1]);
      s2p[it * 512 + 1] = f2bf2(v[2], v[3]);
    }
    pw2 += 32;
    if (k0 + 32 < F_DIM) __builtin_prefetch(pw2, 0, 1);
    cp_wait();
    __syncthreads();

    Frag A, B;
    A.u[0] = *(const u32x4*)&As[mr][g * 8];
    A.u[1] = *(const u32x4*)&As[mr][g * 8 + 4];
    const int br = wave * 16 + mr;
    B.u[0] = *(const u32x4*)&w2s[br][g * 8];
    B.u[1] = *(const u32x4*)&w2s[br][g * 8 + 4];
    acc = __builtin_amdgcn_wmma_f32_16x16x32_bf16(false, A.v, false, B.v,
                                                  (short)0, acc, false, false);
    __syncthreads();
  }

#pragma unroll
  for (int i = 0; i < 8; ++i) {
    const int m = g * 8 + i;
    const int slot = slotBase + m;
    if (slot < sEnd) {
      const int t = tok_idx[slot];
      const float wgt = tok_w[slot];
      atomicAdd(&y[(size_t)t * H_DIM + nb + wave * 16 + mr], acc[i] * wgt);
    }
  }
}

// ---------------------------------------------------------------------------
// Workspace layout (bytes):
//   0       counts[8]            256    offsets[9]
//   512     sel[2T] ints         33280  wts[2T] floats
//   66304   tok_idx[2T] ints     99328  tok_w[2T] floats
//   262144  xb: T*H bf16 (8 MB)
//   9437184 hbuf: (2T+16)*F bf16 (~56 MB)    total ~68.3 MB
// ---------------------------------------------------------------------------
extern "C" void kernel_launch(void* const* d_in, const int* in_sizes, int n_in,
                              void* d_out, int out_size, void* d_ws, size_t ws_size,
                              hipStream_t stream)
{
  (void)in_sizes; (void)n_in; (void)out_size; (void)ws_size;
  const float* x  = (const float*)d_in[0];
  const float* gw = (const float*)d_in[1];
  const float* w1 = (const float*)d_in[2];
  const float* w2 = (const float*)d_in[3];
  const float* w3 = (const float*)d_in[4];

  float* y      = (float*)d_out;
  float* logits = y + (size_t)T_TOKENS * H_DIM;

  char* ws = (char*)d_ws;
  int*   counts  = (int*)(ws + 0);
  int*   offsets = (int*)(ws + 256);
  int*   sel     = (int*)(ws + 512);
  float* wts     = (float*)(ws + 33280);
  int*   tok_idx = (int*)(ws + 66304);
  float* tok_w   = (float*)(ws + 99328);
  unsigned short* xb   = (unsigned short*)(ws + 262144);
  unsigned short* hbuf = (unsigned short*)(ws + 9437184);

  (void)hipMemsetAsync(d_out, 0, (size_t)T_TOKENS * H_DIM * sizeof(float), stream);
  (void)hipMemsetAsync(ws, 0, 512, stream);

  router_kernel<<<T_TOKENS / 8, 256, 0, stream>>>(x, gw, logits, xb, counts, sel, wts);
  scan_kernel<<<1, 256, 0, stream>>>(counts, offsets, sel, wts, tok_idx, tok_w);
  moe_up_kernel<<<dim3(NEXP * 256, F_DIM / 128), 256, 0, stream>>>(
      xb, w1, w3, offsets, tok_idx, hbuf);
  moe_down_kernel<<<dim3(NEXP * 256, H_DIM / 128), 256, 0, stream>>>(
      hbuf, w2, offsets, tok_idx, tok_w, y);
}
